// NomicBertAttention_69080253989152
// MI455X (gfx1250) — matrix operand order repaired and checked by reading, not compiled, not run on hardware
//
#include <hip/hip_runtime.h>
#include <hip/hip_bf16.h>

typedef unsigned short u16;
typedef __attribute__((ext_vector_type(16))) __bf16 v16bf;
typedef __attribute__((ext_vector_type(8)))  float  v8f;
typedef __attribute__((ext_vector_type(4)))  unsigned int u32x4;
typedef __attribute__((ext_vector_type(8)))  unsigned int u32x8;

#define BB  2
#define SS  4096
#define DMM 768
#define HH  12
#define HDD 64

union FragBF { uint4 u[2]; v16bf v; };

__device__ __forceinline__ u16 f2bf(float f) {
  union { float f; unsigned u; } x; x.f = f;
  unsigned r = x.u + 0x7FFFu + ((x.u >> 16) & 1u);
  return (u16)(r >> 16);
}
__device__ __forceinline__ float bf2f(u16 h) {
  union { unsigned u; float f; } x; x.u = ((unsigned)h) << 16;
  return x.f;
}
__device__ __forceinline__ v8f wmma_bf16(const FragBF& a, const FragBF& b, v8f c) {
  return __builtin_amdgcn_wmma_f32_16x16x32_bf16(false, a.v, false, b.v, (short)0, c, false, false);
}

// ---- Tensor Data Mover: 2D tile load Global->LDS (D# groups 0+1 per CDNA5 ISA)
__device__ __forceinline__ void tdm_load_2d(unsigned ldsOff, unsigned long long gaddr,
                                            u32x8 g1) {
  u32x4 g0;
  g0.x = 1u;                                                       // count=1, user mode
  g0.y = ldsOff;                                                   // lds_addr (bytes)
  g0.z = (unsigned)gaddr;                                          // global_addr[31:0]
  g0.w = ((unsigned)(gaddr >> 32) & 0x01FFFFFFu) | 0x80000000u;    // addr[56:32] | type=2
  asm volatile("tensor_load_to_lds %0, %1" :: "s"(g0), "s"(g1) : "memory");
}
__device__ __forceinline__ void wait_tensorcnt0() { asm volatile("s_wait_tensorcnt 0x0" ::: "memory"); }
__device__ __forceinline__ void wait_tensorcnt2() { asm volatile("s_wait_tensorcnt 0x2" ::: "memory"); }
__device__ __forceinline__ void wait_dscnt0()     { asm volatile("s_wait_dscnt 0x0"     ::: "memory"); }

// ---------------------------------------------------------------- convert
__global__ __launch_bounds__(256) void convert_kernel(const float* __restrict__ src,
                                                      u16* __restrict__ dst, int n) {
  int i = blockIdx.x * 256 + threadIdx.x;
  if (i < n) dst[i] = f2bf(src[i]);
}

// ---------------------------------------------------------------- QKV GEMM
__global__ __launch_bounds__(32) void qkv_kernel(
    const u16* __restrict__ Xb,
    const u16* __restrict__ Wqb, const u16* __restrict__ Wkb, const u16* __restrict__ Wvb,
    const float* __restrict__ bq, const float* __restrict__ bk, const float* __restrict__ bv,
    u16* __restrict__ Qr, u16* __restrict__ Kr, u16* __restrict__ Vt) {
  const int lane = threadIdx.x & 31;
  const int kh = lane >> 4, ln = lane & 15;
  const int m0 = blockIdx.x << 4, n0 = blockIdx.y << 4;
  const int which = blockIdx.z;
  const u16* Wb = (which == 0) ? Wqb : (which == 1) ? Wkb : Wvb;
  const float* bias = (which == 0) ? bq : (which == 1) ? bk : bv;

  const u16* xr = Xb + (size_t)(m0 + ln) * DMM;
  const u16* wr = Wb + (size_t)(n0 + ln) * DMM;
  v8f acc = {};
#pragma unroll 4
  for (int k0 = 0; k0 < DMM; k0 += 32) {
    FragBF a, b;
    a.u[0] = *(const uint4*)(xr + k0 + kh * 8);
    a.u[1] = *(const uint4*)(xr + k0 + 16 + kh * 8);
    b.u[0] = *(const uint4*)(wr + k0 + kh * 8);
    b.u[1] = *(const uint4*)(wr + k0 + 16 + kh * 8);
    acc = wmma_bf16(a, b, acc);
  }
  const float bval = bias[n0 + ln];
  const int n = n0 + ln;
  const int h = n >> 6, d = n & 63;
#pragma unroll
  for (int r = 0; r < 8; ++r) {
    const int m = m0 + r + (kh << 3);
    const int bI = m >> 12, s = m & 4095;
    const float v = acc[r] + bval;
    if (which < 2) {
      u16* dst = (which == 0) ? Qr : Kr;
      dst[(((size_t)(bI * HH + h)) * SS + s) * HDD + d] = f2bf(v);
    } else {
      Vt[(((size_t)(bI * HH + h)) * HDD + d) * SS + s] = f2bf(v);
    }
  }
}

// ---------------------------------------------------------------- RoPE (+ fold SCALING into Q)
__global__ __launch_bounds__(256) void rope_kernel(u16* __restrict__ Q, u16* __restrict__ Kv,
                                                   const float* __restrict__ cosT,
                                                   const float* __restrict__ sinT) {
  const int idx = blockIdx.x * 256 + threadIdx.x;
  const int p = idx & 31;
  const int s = (idx >> 5) & 4095;
  const int bh = idx >> 17;
  if (bh >= BB * HH) return;
  const size_t base = ((size_t)bh * SS + s) * HDD;
  const float c = cosT[s * HDD + p];
  const float sn = sinT[s * HDD + p];
  const float q1 = bf2f(Q[base + p]), q2 = bf2f(Q[base + p + 32]);
  Q[base + p]      = f2bf((q1 * c - q2 * sn) * 0.125f);
  Q[base + p + 32] = f2bf((q2 * c + q1 * sn) * 0.125f);
  const float k1 = bf2f(Kv[base + p]), k2 = bf2f(Kv[base + p + 32]);
  Kv[base + p]      = f2bf(k1 * c - k2 * sn);
  Kv[base + p + 32] = f2bf(k2 * c + k1 * sn);
}

// ---------------------------------------------------------------- Flash attention
// One wave: 16 query rows for one (b,h). K/V tiles streamed into LDS by the
// Tensor Data Mover, double-buffered on TENSORcnt; QK^T and PV on WMMA.
__global__ __launch_bounds__(32) void attn_kernel(const u16* __restrict__ Q,
                                                  const u16* __restrict__ K,
                                                  const u16* __restrict__ Vt,
                                                  u16* __restrict__ Ctx) {
  __shared__ __align__(16) u16 Kl[2][32 * 64];   // 32 key rows x 64 dims
  __shared__ __align__(16) u16 Vl[2][64 * 32];   // 64 dims x 32 key cols (Vt layout)
  __shared__ __align__(16) u16 P[16 * 32];       // probs tile

  const int lane = threadIdx.x & 31;
  const int kh = lane >> 4, ln = lane & 15;
  const int bh = blockIdx.y;
  const int s0 = blockIdx.x << 4;
  const u16* Qp = Q + ((size_t)bh * SS + s0) * HDD;
  const u16* Kp = K + (size_t)bh * SS * HDD;
  const u16* Vp = Vt + (size_t)bh * HDD * SS;

  FragBF aq0, aq1;
  {
    const u16* qr = Qp + ln * HDD;
    aq0.u[0] = *(const uint4*)(qr + kh * 8);
    aq0.u[1] = *(const uint4*)(qr + 16 + kh * 8);
    aq1.u[0] = *(const uint4*)(qr + 32 + kh * 8);
    aq1.u[1] = *(const uint4*)(qr + 48 + kh * 8);
  }
  v8f o0 = {}, o1 = {}, o2 = {}, o3 = {};
  float mrow[8], lrow[8];
#pragma unroll
  for (int r = 0; r < 8; ++r) { mrow[r] = -3.0e38f; lrow[r] = 0.f; }

  // D# group1 descriptors (data_size=2B; 2D tensor; tile K:64x32, V:32x64)
  const u32x8 dK = { 0x00010000u, (64u << 16),   (4096u << 16), (64u << 16), 32u, 64u,   0u, 0u };
  const u32x8 dV = { 0x00010000u, (4096u << 16), (64u << 16),   (32u << 16), 64u, 4096u, 0u, 0u };
  const unsigned long long kBase = (unsigned long long)(const char*)Kp;
  const unsigned long long vBase = (unsigned long long)(const char*)Vp;
  const unsigned ldsK0 = (unsigned)(size_t)&Kl[0][0];
  const unsigned ldsK1 = (unsigned)(size_t)&Kl[1][0];
  const unsigned ldsV0 = (unsigned)(size_t)&Vl[0][0];
  const unsigned ldsV1 = (unsigned)(size_t)&Vl[1][0];

  // preload tile 0
  tdm_load_2d(ldsK0, kBase, dK);
  tdm_load_2d(ldsV0, vBase, dV);
  int cur = 0;

  for (int t0 = 0; t0 < SS; t0 += 32) {
    if (t0 + 32 < SS) {
      wait_dscnt0();                                   // WAR: prior reads of the refill buffer
      const unsigned long long nk = kBase + (unsigned long long)(t0 + 32) * (HDD * 2);
      const unsigned long long nv = vBase + (unsigned long long)(t0 + 32) * 2;
      tdm_load_2d(cur ? ldsK0 : ldsK1, nk, dK);
      tdm_load_2d(cur ? ldsV0 : ldsV1, nv, dV);
      wait_tensorcnt2();                               // current tile's pair complete
    } else {
      wait_tensorcnt0();
    }
    const u16* Kc = &Kl[cur][0];
    const u16* Vc = &Vl[cur][0];

    FragBF bk00, bk01, bk10, bk11;
    bk00.u[0] = *(const uint4*)(Kc + ln * 64 + kh * 8);
    bk00.u[1] = *(const uint4*)(Kc + ln * 64 + 16 + kh * 8);
    bk01.u[0] = *(const uint4*)(Kc + ln * 64 + 32 + kh * 8);
    bk01.u[1] = *(const uint4*)(Kc + ln * 64 + 48 + kh * 8);
    bk10.u[0] = *(const uint4*)(Kc + (16 + ln) * 64 + kh * 8);
    bk10.u[1] = *(const uint4*)(Kc + (16 + ln) * 64 + 16 + kh * 8);
    bk11.u[0] = *(const uint4*)(Kc + (16 + ln) * 64 + 32 + kh * 8);
    bk11.u[1] = *(const uint4*)(Kc + (16 + ln) * 64 + 48 + kh * 8);

    v8f sc0 = {}, sc1 = {};
    sc0 = wmma_bf16(aq0, bk00, sc0);
    sc0 = wmma_bf16(aq1, bk01, sc0);
    sc1 = wmma_bf16(aq0, bk10, sc1);
    sc1 = wmma_bf16(aq1, bk11, sc1);

    float p0[8], p1[8], alph[8];
#pragma unroll
    for (int r = 0; r < 8; ++r) {
      const float a = sc0[r] * 0.125f;
      const float b = sc1[r] * 0.125f;
      float mx = fmaxf(a, b);
      mx = fmaxf(mx, __shfl_xor(mx, 1, 16));
      mx = fmaxf(mx, __shfl_xor(mx, 2, 16));
      mx = fmaxf(mx, __shfl_xor(mx, 4, 16));
      mx = fmaxf(mx, __shfl_xor(mx, 8, 16));
      const float mnew = fmaxf(mrow[r], mx);
      const float al = __expf(mrow[r] - mnew);
      const float e0 = __expf(a - mnew);
      const float e1 = __expf(b - mnew);
      float rs = e0 + e1;
      rs += __shfl_xor(rs, 1, 16);
      rs += __shfl_xor(rs, 2, 16);
      rs += __shfl_xor(rs, 4, 16);
      rs += __shfl_xor(rs, 8, 16);
      lrow[r] = lrow[r] * al + rs;
      mrow[r] = mnew;
      p0[r] = e0; p1[r] = e1; alph[r] = al;
    }
#pragma unroll
    for (int r = 0; r < 8; ++r) {
      const int row = r + (kh << 3);
      P[row * 32 + ln]      = f2bf(p0[r]);
      P[row * 32 + 16 + ln] = f2bf(p1[r]);
    }
    __syncthreads();
    FragBF ap;
    ap.u[0] = *(const uint4*)(&P[ln * 32 + kh * 8]);
    ap.u[1] = *(const uint4*)(&P[ln * 32 + 16 + kh * 8]);
#pragma unroll
    for (int r = 0; r < 8; ++r) {
      o0[r] *= alph[r]; o1[r] *= alph[r]; o2[r] *= alph[r]; o3[r] *= alph[r];
    }
    FragBF bv;
    bv.u[0] = *(const uint4*)(Vc + (0 * 16 + ln) * 32 + kh * 8);
    bv.u[1] = *(const uint4*)(Vc + (0 * 16 + ln) * 32 + 16 + kh * 8);
    o0 = wmma_bf16(ap, bv, o0);
    bv.u[0] = *(const uint4*)(Vc + (1 * 16 + ln) * 32 + kh * 8);
    bv.u[1] = *(const uint4*)(Vc + (1 * 16 + ln) * 32 + 16 + kh * 8);
    o1 = wmma_bf16(ap, bv, o1);
    bv.u[0] = *(const uint4*)(Vc + (2 * 16 + ln) * 32 + kh * 8);
    bv.u[1] = *(const uint4*)(Vc + (2 * 16 + ln) * 32 + 16 + kh * 8);
    o2 = wmma_bf16(ap, bv, o2);
    bv.u[0] = *(const uint4*)(Vc + (3 * 16 + ln) * 32 + kh * 8);
    bv.u[1] = *(const uint4*)(Vc + (3 * 16 + ln) * 32 + 16 + kh * 8);
    o3 = wmma_bf16(ap, bv, o3);
    __syncthreads();
    cur ^= 1;
  }
  const int bI = bh / HH, h = bh % HH;
#pragma unroll
  for (int r = 0; r < 8; ++r) {
    const int s = s0 + r + (kh << 3);
    u16* cr = Ctx + ((size_t)(bI * SS + s)) * DMM + h * HDD + ln;
    const float inv = 1.0f / lrow[r];
    cr[0]  = f2bf(o0[r] * inv);
    cr[16] = f2bf(o1[r] * inv);
    cr[32] = f2bf(o2[r] * inv);
    cr[48] = f2bf(o3[r] * inv);
  }
}

// ---------------------------------------------------------------- out projection + residual
__global__ __launch_bounds__(32) void outproj_kernel(const u16* __restrict__ Ctx,
                                                     const u16* __restrict__ Wob,
                                                     const float* __restrict__ hidden,
                                                     float* __restrict__ preLN) {
  const int lane = threadIdx.x & 31;
  const int kh = lane >> 4, ln = lane & 15;
  const int m0 = blockIdx.x << 4, n0 = blockIdx.y << 4;
  const u16* xr = Ctx + (size_t)(m0 + ln) * DMM;
  const u16* wr = Wob + (size_t)(n0 + ln) * DMM;
  v8f acc = {};
#pragma unroll 4
  for (int k0 = 0; k0 < DMM; k0 += 32) {
    FragBF a, b;
    a.u[0] = *(const uint4*)(xr + k0 + kh * 8);
    a.u[1] = *(const uint4*)(xr + k0 + 16 + kh * 8);
    b.u[0] = *(const uint4*)(wr + k0 + kh * 8);
    b.u[1] = *(const uint4*)(wr + k0 + 16 + kh * 8);
    acc = wmma_bf16(a, b, acc);
  }
#pragma unroll
  for (int r = 0; r < 8; ++r) {
    const int m = m0 + r + (kh << 3);
    const int n = n0 + ln;
    preLN[(size_t)m * DMM + n] = acc[r] + hidden[(size_t)m * DMM + n];
  }
}

// ---------------------------------------------------------------- LayerNorm
__global__ __launch_bounds__(256) void ln_kernel(const float* __restrict__ x,
                                                 const float* __restrict__ g,
                                                 const float* __restrict__ b,
                                                 float* __restrict__ out) {
  const int row = blockIdx.x;
  const float* xr = x + (size_t)row * DMM;
  const int t = threadIdx.x;
  float s = 0.f, sq = 0.f;
  for (int i = t; i < DMM; i += 256) { const float v = xr[i]; s += v; sq += v * v; }
  __shared__ float sh_s[256], sh_q[256];
  sh_s[t] = s; sh_q[t] = sq; __syncthreads();
  for (int off = 128; off > 0; off >>= 1) {
    if (t < off) { sh_s[t] += sh_s[t + off]; sh_q[t] += sh_q[t + off]; }
    __syncthreads();
  }
  const float mean = sh_s[0] * (1.0f / DMM);
  const float var  = sh_q[0] * (1.0f / DMM) - mean * mean;
  const float inv  = rsqrtf(var + 1e-12f);
  for (int i = t; i < DMM; i += 256) {
    const float v = xr[i];
    out[(size_t)row * DMM + i] = (v - mean) * inv * g[i] + b[i];
  }
}

// ---------------------------------------------------------------- launch
extern "C" void kernel_launch(void* const* d_in, const int* in_sizes, int n_in,
                              void* d_out, int out_size, void* d_ws, size_t ws_size,
                              hipStream_t stream) {
  (void)in_sizes; (void)n_in; (void)out_size; (void)ws_size;
  const float* hidden = (const float*)d_in[0];
  const float* cosT   = (const float*)d_in[1];
  const float* sinT   = (const float*)d_in[2];
  const float* Wq     = (const float*)d_in[3];
  const float* bq     = (const float*)d_in[4];
  const float* Wk     = (const float*)d_in[5];
  const float* bk     = (const float*)d_in[6];
  const float* Wv     = (const float*)d_in[7];
  const float* bv     = (const float*)d_in[8];
  const float* Wo     = (const float*)d_in[9];
  const float* lng    = (const float*)d_in[10];
  const float* lnb    = (const float*)d_in[11];
  float* out = (float*)d_out;

  char* ws = (char*)d_ws;
  const size_t nX = (size_t)BB * SS * DMM;
  const size_t nW = (size_t)DMM * DMM;
  const size_t nQ = (size_t)BB * HH * SS * HDD;
  u16* Xb  = (u16*)ws;  ws += nX * 2;
  u16* Wqb = (u16*)ws;  ws += nW * 2;
  u16* Wkb = (u16*)ws;  ws += nW * 2;
  u16* Wvb = (u16*)ws;  ws += nW * 2;
  u16* Wob = (u16*)ws;  ws += nW * 2;
  u16* Qr  = (u16*)ws;  ws += nQ * 2;
  u16* Kr  = (u16*)ws;  ws += nQ * 2;
  u16* Vt  = (u16*)ws;  ws += nQ * 2;
  u16* Ctx = (u16*)ws;  ws += nX * 2;
  float* preLN = (float*)ws;

  convert_kernel<<<dim3((unsigned)((nX + 255) / 256)), 256, 0, stream>>>(hidden, Xb, (int)nX);
  convert_kernel<<<dim3((unsigned)((nW + 255) / 256)), 256, 0, stream>>>(Wq, Wqb, (int)nW);
  convert_kernel<<<dim3((unsigned)((nW + 255) / 256)), 256, 0, stream>>>(Wk, Wkb, (int)nW);
  convert_kernel<<<dim3((unsigned)((nW + 255) / 256)), 256, 0, stream>>>(Wv, Wvb, (int)nW);
  convert_kernel<<<dim3((unsigned)((nW + 255) / 256)), 256, 0, stream>>>(Wo, Wob, (int)nW);

  qkv_kernel<<<dim3(512, 48, 3), 32, 0, stream>>>(Xb, Wqb, Wkb, Wvb, bq, bk, bv, Qr, Kr, Vt);
  rope_kernel<<<dim3((BB * HH * SS * 32) / 256), 256, 0, stream>>>(Qr, Kr, cosT, sinT);
  attn_kernel<<<dim3(SS / 16, BB * HH), 32, 0, stream>>>(Qr, Kr, Vt, Ctx);
  outproj_kernel<<<dim3(512, 48), 32, 0, stream>>>(Ctx, Wob, hidden, preLN);
  ln_kernel<<<dim3(BB * SS), 256, 0, stream>>>(preLN, lng, lnb, out);
}
